// Attention_82446192214489
// MI455X (gfx1250) — matrix operand order repaired and checked by reading, not compiled, not run on hardware
//
#include <hip/hip_runtime.h>

typedef _Float16 v16h __attribute__((ext_vector_type(16)));
typedef _Float16 v8h  __attribute__((ext_vector_type(8)));
typedef float    v8f  __attribute__((ext_vector_type(8)));
typedef unsigned int v4u __attribute__((ext_vector_type(4)));
typedef int v4i __attribute__((ext_vector_type(4)));
typedef int v8i __attribute__((ext_vector_type(8)));

#define AS1 __attribute__((address_space(1)))
#define AS3 __attribute__((address_space(3)))

namespace {
constexpr int Cdim = 384;   // channels / model dim
constexpr int NHEAD = 6;
constexpr int HD    = 64;   // head dim
constexpr int Bsz   = 64;   // batch
constexpr int Himg  = 28, Wimg = 28;
constexpr int TQ    = 784;  // q tokens
constexpr int MQ    = 832;  // q tokens padded to 13*64
constexpr int TK    = 169;  // kv tokens (13x13)
constexpr int MK    = 192;  // kv tokens padded to 3*64
constexpr float EPSBN = 1e-3f;
constexpr int PSTR  = 200;  // LDS row stride (halves) for P tiles
}

#if __has_builtin(__builtin_amdgcn_tensor_load_to_lds)
#define HAVE_TDM 1
#else
#define HAVE_TDM 0
#endif

__device__ __forceinline__ unsigned int lds_off32(const void* p) {
  return (unsigned int)(uintptr_t)p;  // generic LDS addr low 32 bits = LDS offset
}

// ---- async global -> LDS b128 (per-lane, ASYNCcnt) ---------------------------
__device__ __forceinline__ void async_copy_b128(const _Float16* g, _Float16* l) {
#if __has_builtin(__builtin_amdgcn_global_load_async_to_lds_b128)
  __builtin_amdgcn_global_load_async_to_lds_b128(
      (AS1 v4i*)(uintptr_t)g, (AS3 v4i*)lds_off32(l), 0, 0);
#else
  asm volatile("global_load_async_to_lds_b128 %0, %1, off"
               :: "v"(lds_off32(l)), "v"(g) : "memory");
#endif
}
__device__ __forceinline__ void async_wait0() {
#if __has_builtin(__builtin_amdgcn_s_wait_asynccnt)
  __builtin_amdgcn_s_wait_asynccnt(0);
#else
  asm volatile("s_wait_asynccnt 0x0" ::: "memory");
#endif
}

// ---- TDM: 2-D tile (rows x row_len u16 elems, global row stride in elems) ----
__device__ __forceinline__ void tdm_load_2d(const _Float16* gsrc,
                                            unsigned int lds_byte_off,
                                            unsigned int rows,
                                            unsigned int row_len,
                                            unsigned int row_stride_elems) {
#if HAVE_TDM
  unsigned long long ga = (unsigned long long)(uintptr_t)gsrc;
  v4u g0 = {0u, 0u, 0u, 0u};
  g0[0] = 1u;                                   // count=1 valid descriptor
  g0[1] = lds_byte_off;                         // D#.lds_addr
  g0[2] = (unsigned int)ga;                     // global_addr[31:0]
  g0[3] = (unsigned int)((ga >> 32) & 0x01FFFFFFu) | (2u << 30);  // addr[56:32]|type=2
  v8i g1 = {0, 0, 0, 0, 0, 0, 0, 0};
  g1[0] = (int)(1u << 16);                      // data_size=1 -> 2-byte elems
  g1[1] = (int)(row_len << 16);                 // tensor_dim0[15:0] @bits63:48
  g1[2] = (int)(rows << 16);                    // tensor_dim1[15:0] @bits95:80
  g1[3] = (int)(row_len << 16);                 // tile_dim0 @bits127:112
  g1[4] = (int)rows;                            // tile_dim1 (tile_dim2=0)
  g1[5] = (int)row_stride_elems;                // tensor_dim0_stride[31:0]
  v4i z4 = {0, 0, 0, 0};
#if defined(__clang_major__) && __clang_major__ >= 23
  v8i z8 = {0, 0, 0, 0, 0, 0, 0, 0};
  __builtin_amdgcn_tensor_load_to_lds(g0, g1, z4, z4, z8, 0);
#else
  __builtin_amdgcn_tensor_load_to_lds(g0, g1, z4, z4, 0);
#endif
#else
  (void)gsrc; (void)lds_byte_off; (void)rows; (void)row_len; (void)row_stride_elems;
#endif
}
__device__ __forceinline__ void tensor_wait0() {
#if __has_builtin(__builtin_amdgcn_s_wait_tensorcnt)
  __builtin_amdgcn_s_wait_tensorcnt(0);
#else
  asm volatile("s_wait_tensorcnt 0x0" ::: "memory");
#endif
}

// ---------------- WMMA fragment helpers (wave32, 16x16x32 f16) ----------------
// A (16x32, M x K): lane L holds row M=L&15; lanes 0-15 K={k..k+7, k+16..k+23},
// lanes 16-31 K={k+8..k+15, k+24..k+31}  (ISA 7.12.2 16-bit A layout)
__device__ __forceinline__ v16h load_a_frag(const _Float16* p, int lane) {
  const int ka = (lane >= 16) ? 8 : 0;
  v8h lo = *reinterpret_cast<const v8h*>(p + ka);
  v8h hi = *reinterpret_cast<const v8h*>(p + ka + 16);
  v16h r;
#pragma unroll
  for (int i = 0; i < 8; ++i) { r[i] = lo[i]; r[i + 8] = hi[i]; }
  return r;
}
// B (32x16, K x N): lane L holds column N=L&15; lanes 0-15 K={k..k+7, k+16..k+23},
// lanes 16-31 K={k+8..k+15, k+24..k+31} (same K order as A).
__device__ __forceinline__ v16h load_b_frag(const _Float16* p, int lane) {
  const int kb = (lane >= 16) ? 8 : 0;
  v8h lo = *reinterpret_cast<const v8h*>(p + kb);
  v8h hi = *reinterpret_cast<const v8h*>(p + kb + 16);
  v16h r;
#pragma unroll
  for (int i = 0; i < 8; ++i) { r[i] = lo[i]; r[i + 8] = hi[i]; }
  return r;
}
__device__ __forceinline__ v8f wmma32(v16h a, v16h b, v8f c) {
  return __builtin_amdgcn_wmma_f32_16x16x32_f16(false, a, false, b, (short)0, c,
                                                false, false);
}

// ---------------- Stage 0: weight convert+transpose W[k][n] -> Wt[n][k] f16 ----
__global__ void wt_kernel(const float* __restrict__ W, _Float16* __restrict__ Wt) {
  int idx = blockIdx.x * blockDim.x + threadIdx.x;
  if (idx >= Cdim * Cdim) return;
  int n = idx / Cdim, k = idx % Cdim;
  Wt[idx] = (_Float16)W[k * Cdim + n];
}

// ---------------- Stage 1a: depthwise 3x3 s1 SAME + BN -> f16 (padded to MQ) --
__global__ void dwbn_q_kernel(const float* __restrict__ x,
                              const float* __restrict__ wdw,
                              const float* __restrict__ gam,
                              const float* __restrict__ bet,
                              const float* __restrict__ mea,
                              const float* __restrict__ va,
                              _Float16* __restrict__ out) {
  int idx = blockIdx.x * blockDim.x + threadIdx.x;
  if (idx >= Bsz * MQ * Cdim) return;
  int c = idx % Cdim;
  int t = (idx / Cdim) % MQ;
  int b = idx / (Cdim * MQ);
  if (t >= TQ) { out[idx] = (_Float16)0.f; return; }
  int i = t / Wimg, j = t % Wimg;
  float acc = 0.f;
#pragma unroll
  for (int di = 0; di < 3; ++di)
#pragma unroll
    for (int dj = 0; dj < 3; ++dj) {
      int ii = i + di - 1, jj = j + dj - 1;
      if (ii >= 0 && ii < Himg && jj >= 0 && jj < Wimg)
        acc += x[((size_t)b * TQ + ii * Wimg + jj) * Cdim + c] *
               wdw[(di * 3 + dj) * Cdim + c];
    }
  float sc = gam[c] * rsqrtf(va[c] + EPSBN);
  out[idx] = (_Float16)(sc * (acc - mea[c]) + bet[c]);
}

// ---------------- Stage 1b: depthwise 3x3 s2 VALID + BN for k & v -------------
__global__ void dwbn_kv_kernel(const float* __restrict__ x,
                               const float* __restrict__ wkd, const float* __restrict__ kg,
                               const float* __restrict__ kbt, const float* __restrict__ kmn,
                               const float* __restrict__ kvr,
                               const float* __restrict__ wvd, const float* __restrict__ vg,
                               const float* __restrict__ vbt, const float* __restrict__ vmn,
                               const float* __restrict__ vvr,
                               _Float16* __restrict__ outk, _Float16* __restrict__ outv) {
  int idx = blockIdx.x * blockDim.x + threadIdx.x;
  if (idx >= Bsz * MK * Cdim) return;
  int c = idx % Cdim;
  int t = (idx / Cdim) % MK;
  int b = idx / (Cdim * MK);
  if (t >= TK) { outk[idx] = (_Float16)0.f; outv[idx] = (_Float16)0.f; return; }
  int oi = t / 13, oj = t % 13;
  float ak = 0.f, av = 0.f;
#pragma unroll
  for (int di = 0; di < 3; ++di)
#pragma unroll
    for (int dj = 0; dj < 3; ++dj) {
      int ii = 2 * oi + di, jj = 2 * oj + dj;  // always in bounds (<=26)
      float xv = x[((size_t)b * TQ + ii * Wimg + jj) * Cdim + c];
      ak += xv * wkd[(di * 3 + dj) * Cdim + c];
      av += xv * wvd[(di * 3 + dj) * Cdim + c];
    }
  float sk = kg[c] * rsqrtf(kvr[c] + EPSBN);
  float sv = vg[c] * rsqrtf(vvr[c] + EPSBN);
  outk[idx] = (_Float16)(sk * (ak - kmn[c]) + kbt[c]);
  outv[idx] = (_Float16)(sv * (av - vmn[c]) + vbt[c]);
}

// ---------------- Stage 2: WMMA GEMM  D[b] = A[b] @ Wt^T  (K=384) -------------
// A-panel (64x384) staged via async b128 copies; B-panel (64x384) via TDM.
__global__ void gemm_proj_kernel(const _Float16* __restrict__ A,
                                 const _Float16* __restrict__ Wt,
                                 _Float16* __restrict__ D, int M, int transposed) {
  __shared__ _Float16 As[64 * Cdim];  // 48KB
  __shared__ _Float16 Bs[64 * Cdim];  // 48KB
  const int b = blockIdx.z;
  const int lane = threadIdx.x & 31;
  const int wave = threadIdx.x >> 5;           // 4 waves, 2x2 of 32x32 tiles
  const int lr = lane & 15;

  // stage A panel: contiguous 64 rows of A
  const _Float16* asrc = A + ((size_t)b * M + (size_t)blockIdx.x * 64) * Cdim;
  for (int i = threadIdx.x; i < (64 * Cdim) / 8; i += 128)
    async_copy_b128(asrc + (size_t)i * 8, &As[i * 8]);
  // stage B panel via TDM (one descriptor, wave 0)
#if HAVE_TDM
  if (wave == 0)
    tdm_load_2d(Wt + (size_t)blockIdx.y * 64 * Cdim, lds_off32(Bs), 64, Cdim, Cdim);
#else
  {
    const _Float16* wsrc = Wt + (size_t)blockIdx.y * 64 * Cdim;
    for (int i = threadIdx.x; i < (64 * Cdim) / 8; i += 128)
      async_copy_b128(wsrc + (size_t)i * 8, &Bs[i * 8]);
  }
#endif
  async_wait0();
#if HAVE_TDM
  if (wave == 0) tensor_wait0();
#endif
  __syncthreads();

  const _Float16* a0p = As + (size_t)((wave >> 1) * 32 + lr) * Cdim;
  const _Float16* a1p = a0p + (size_t)16 * Cdim;
  const _Float16* b0p = Bs + (size_t)((wave & 1) * 32 + lr) * Cdim;
  const _Float16* b1p = b0p + (size_t)16 * Cdim;
  v8f acc00 = {}, acc01 = {}, acc10 = {}, acc11 = {};
  for (int kk = 0; kk < Cdim; kk += 32) {
    v16h a0 = load_a_frag(a0p + kk, lane);
    v16h a1 = load_a_frag(a1p + kk, lane);
    v16h w0 = load_b_frag(b0p + kk, lane);
    v16h w1 = load_b_frag(b1p + kk, lane);
    acc00 = wmma32(a0, w0, acc00);
    acc01 = wmma32(a0, w1, acc01);
    acc10 = wmma32(a1, w0, acc10);
    acc11 = wmma32(a1, w1, acc11);
  }
  const int m_base = blockIdx.x * 64 + (wave >> 1) * 32;
  const int n_base = blockIdx.y * 64 + (wave & 1) * 32;
  const int hi8 = (lane >= 16) ? 8 : 0;
#pragma unroll
  for (int r = 0; r < 8; ++r) {
    int row0 = m_base + r + hi8, row1 = row0 + 16;
    int col0 = n_base + lr,      col1 = col0 + 16;
    if (!transposed) {
      _Float16* Db = D + (size_t)b * M * Cdim;
      Db[(size_t)row0 * Cdim + col0] = (_Float16)acc00[r];
      Db[(size_t)row0 * Cdim + col1] = (_Float16)acc01[r];
      Db[(size_t)row1 * Cdim + col0] = (_Float16)acc10[r];
      Db[(size_t)row1 * Cdim + col1] = (_Float16)acc11[r];
    } else {  // D[b][col][row], row dim = MK (for V: PV B-frags contiguous)
      _Float16* Db = D + (size_t)b * Cdim * MK;
      Db[(size_t)col0 * MK + row0] = (_Float16)acc00[r];
      Db[(size_t)col1 * MK + row0] = (_Float16)acc01[r];
      Db[(size_t)col0 * MK + row1] = (_Float16)acc10[r];
      Db[(size_t)col1 * MK + row1] = (_Float16)acc11[r];
    }
  }
}

// ---------------- Stage 3: fused attention per (batch, head, 64-row tile) -----
// Q: [B][MQ][384], Kp: [B][MK][384], Vt: [B][384][MK], O: [B][MQ][384] (all f16)
__global__ void attn_kernel(const _Float16* __restrict__ Q,
                            const _Float16* __restrict__ Kp,
                            const _Float16* __restrict__ Vt,
                            _Float16* __restrict__ O) {
  __shared__ _Float16 Ks[MK * HD];          // 24KB  K panel [j][d]
  __shared__ _Float16 Vs[HD * MK];          // 24KB  V panel [d][t]
  __shared__ _Float16 plds[4 * 16 * PSTR];  // 25KB  per-wave P tiles
  const int b = blockIdx.z;
  const int h = blockIdx.y;
  const int lane = threadIdx.x & 31;
  const int wave = threadIdx.x >> 5;
  const int lr = lane & 15;
  const int hi8 = (lane >= 16) ? 8 : 0;
  const int mrow = blockIdx.x * 64 + wave * 16;  // 16 q-rows per wave

  // stage K/V panels for this (b,h)
  const _Float16* ksrc = Kp + (size_t)b * MK * Cdim + h * HD;          // stride Cdim
  const _Float16* vsrc = Vt + ((size_t)b * Cdim + h * HD) * MK;        // stride MK
#if HAVE_TDM
  if (wave == 0) {
    tdm_load_2d(ksrc, lds_off32(Ks), MK, HD, Cdim);
    tdm_load_2d(vsrc, lds_off32(Vs), HD, MK, MK);
    tensor_wait0();
  }
#else
  for (int i = threadIdx.x; i < (MK * HD) / 8; i += 128) {
    int j = i >> 3, d8 = (i & 7) * 8;  // 8 b128-chunks per 64-elem row
    async_copy_b128(ksrc + (size_t)j * Cdim + d8, &Ks[j * HD + d8]);
  }
  for (int i = threadIdx.x; i < (HD * MK) / 8; i += 128) {
    int d = i / 24, t8 = (i % 24) * 8;  // 24 b128-chunks per 192-elem row
    async_copy_b128(vsrc + (size_t)d * MK + t8, &Vs[d * MK + t8]);
  }
  async_wait0();
#endif
  __syncthreads();

  const _Float16* qrow = Q + ((size_t)b * MQ + (mrow + lr)) * Cdim + h * HD;

  // S = Q Kt over all 192 (padded) keys: 12 n-tiles of 16, K-loop over d=64
  v8f s[12];
#pragma unroll
  for (int t = 0; t < 12; ++t) {
    v8f acc = {};
#pragma unroll
    for (int kk = 0; kk < HD; kk += 32) {
      v16h a  = load_a_frag(qrow + kk, lane);
      v16h bm = load_b_frag(&Ks[(t * 16 + lr) * HD + kk], lane);
      acc = wmma32(a, bm, acc);
    }
    s[t] = acc;
  }

  // scale + mask + row softmax. Row m lives in VGPR r across one 16-lane half,
  // so xor-shuffles 1,2,4,8 reduce exactly within the owning half-wave.
  const float scale = 0.051031036307982884f;  // 384^-0.5 (dim_out^-0.5)
  float rmax[8], rsum[8], rinv[8];
#pragma unroll
  for (int r = 0; r < 8; ++r) rmax[r] = -3.4e38f;
#pragma unroll
  for (int t = 0; t < 12; ++t) {
    bool oob = (t * 16 + lr) >= TK;
#pragma unroll
    for (int r = 0; r < 8; ++r) {
      float v = s[t][r] * scale;
      v = oob ? -3.4e38f : v;
      s[t][r] = v;
      rmax[r] = fmaxf(rmax[r], v);
    }
  }
#pragma unroll
  for (int r = 0; r < 8; ++r) {
    for (int off = 1; off <= 8; off <<= 1)
      rmax[r] = fmaxf(rmax[r], __shfl_xor(rmax[r], off, 32));
    rsum[r] = 0.f;
  }
#pragma unroll
  for (int t = 0; t < 12; ++t)
#pragma unroll
    for (int r = 0; r < 8; ++r) {
      float e = __expf(s[t][r] - rmax[r]);
      s[t][r] = e;
      rsum[r] += e;
    }
#pragma unroll
  for (int r = 0; r < 8; ++r) {
    for (int off = 1; off <= 8; off <<= 1)
      rsum[r] += __shfl_xor(rsum[r], off, 32);
    rinv[r] = 1.0f / rsum[r];
  }

  // write normalized P to LDS (C-layout -> row-major) for A-fragment reloads
  _Float16* pw = plds + wave * 16 * PSTR;
#pragma unroll
  for (int t = 0; t < 12; ++t)
#pragma unroll
    for (int r = 0; r < 8; ++r)
      pw[(r + hi8) * PSTR + t * 16 + lr] = (_Float16)(s[t][r] * rinv[r]);
  __syncthreads();

  // O = P @ V : M=16 rows, N=64 (4 tiles), K=192 (6 steps)
  const _Float16* prow = plds + wave * 16 * PSTR + lr * PSTR;
#pragma unroll
  for (int dt = 0; dt < 4; ++dt) {
    v8f acc = {};
#pragma unroll
    for (int kk = 0; kk < MK; kk += 32) {
      v16h a  = load_a_frag(prow + kk, lane);
      v16h bm = load_b_frag(&Vs[(dt * 16 + lr) * MK + kk], lane);
      acc = wmma32(a, bm, acc);
    }
    int col = h * HD + dt * 16 + lr;
    size_t obase = (size_t)b * MQ * Cdim;
#pragma unroll
    for (int r = 0; r < 8; ++r)
      O[obase + (size_t)(mrow + hi8 + r) * Cdim + col] = (_Float16)acc[r];
  }
}

// ---------------- Stage 4: out = op @ Wo + bo (f32 store, bounds t<784) -------
__global__ void gemm_out_kernel(const _Float16* __restrict__ A,
                                const _Float16* __restrict__ Wt,
                                const float* __restrict__ bias,
                                float* __restrict__ out) {
  __shared__ _Float16 As[64 * Cdim];
  __shared__ _Float16 Bs[64 * Cdim];
  const int b = blockIdx.z;
  const int lane = threadIdx.x & 31;
  const int wave = threadIdx.x >> 5;
  const int lr = lane & 15;

  const _Float16* asrc = A + ((size_t)b * MQ + (size_t)blockIdx.x * 64) * Cdim;
  for (int i = threadIdx.x; i < (64 * Cdim) / 8; i += 128)
    async_copy_b128(asrc + (size_t)i * 8, &As[i * 8]);
#if HAVE_TDM
  if (wave == 0)
    tdm_load_2d(Wt + (size_t)blockIdx.y * 64 * Cdim, lds_off32(Bs), 64, Cdim, Cdim);
#else
  {
    const _Float16* wsrc = Wt + (size_t)blockIdx.y * 64 * Cdim;
    for (int i = threadIdx.x; i < (64 * Cdim) / 8; i += 128)
      async_copy_b128(wsrc + (size_t)i * 8, &Bs[i * 8]);
  }
#endif
  async_wait0();
#if HAVE_TDM
  if (wave == 0) tensor_wait0();
#endif
  __syncthreads();

  const _Float16* a0p = As + (size_t)((wave >> 1) * 32 + lr) * Cdim;
  const _Float16* a1p = a0p + (size_t)16 * Cdim;
  const _Float16* b0p = Bs + (size_t)((wave & 1) * 32 + lr) * Cdim;
  const _Float16* b1p = b0p + (size_t)16 * Cdim;
  v8f acc00 = {}, acc01 = {}, acc10 = {}, acc11 = {};
  for (int kk = 0; kk < Cdim; kk += 32) {
    v16h a0 = load_a_frag(a0p + kk, lane);
    v16h a1 = load_a_frag(a1p + kk, lane);
    v16h w0 = load_b_frag(b0p + kk, lane);
    v16h w1 = load_b_frag(b1p + kk, lane);
    acc00 = wmma32(a0, w0, acc00);
    acc01 = wmma32(a0, w1, acc01);
    acc10 = wmma32(a1, w0, acc10);
    acc11 = wmma32(a1, w1, acc11);
  }
  const int m_base = blockIdx.x * 64 + (wave >> 1) * 32;
  const int n_base = blockIdx.y * 64 + (wave & 1) * 32;
  const int hi8 = (lane >= 16) ? 8 : 0;
  float bc0 = bias[n_base + lr], bc1 = bias[n_base + lr + 16];
#pragma unroll
  for (int r = 0; r < 8; ++r) {
    int row0 = m_base + r + hi8, row1 = row0 + 16;
    int col0 = n_base + lr,      col1 = col0 + 16;
    if (row0 < TQ) {
      out[((size_t)b * TQ + row0) * Cdim + col0] = acc00[r] + bc0;
      out[((size_t)b * TQ + row0) * Cdim + col1] = acc01[r] + bc1;
    }
    if (row1 < TQ) {
      out[((size_t)b * TQ + row1) * Cdim + col0] = acc10[r] + bc0;
      out[((size_t)b * TQ + row1) * Cdim + col1] = acc11[r] + bc1;
    }
  }
}

// ---------------- launch ------------------------------------------------------
extern "C" void kernel_launch(void* const* d_in, const int* in_sizes, int n_in,
                              void* d_out, int out_size, void* d_ws, size_t ws_size,
                              hipStream_t stream) {
  (void)in_sizes; (void)n_in; (void)out_size; (void)ws_size;
  const float* x     = (const float*)d_in[0];
  const float* wq_dw = (const float*)d_in[1];
  const float* qg = (const float*)d_in[2];
  const float* qb = (const float*)d_in[3];
  const float* qm = (const float*)d_in[4];
  const float* qv = (const float*)d_in[5];
  const float* wk_dw = (const float*)d_in[6];
  const float* kg = (const float*)d_in[7];
  const float* kb = (const float*)d_in[8];
  const float* km = (const float*)d_in[9];
  const float* kv = (const float*)d_in[10];
  const float* wv_dw = (const float*)d_in[11];
  const float* vg = (const float*)d_in[12];
  const float* vb = (const float*)d_in[13];
  const float* vm = (const float*)d_in[14];
  const float* vv = (const float*)d_in[15];
  const float* Wq = (const float*)d_in[16];
  const float* Wk = (const float*)d_in[17];
  const float* Wv = (const float*)d_in[18];
  const float* Wo = (const float*)d_in[19];
  const float* bo = (const float*)d_in[20];

  // workspace carve-up (all f16; ~162 MB total)
  char* ws = (char*)d_ws;
  size_t off = 0;
  auto carve = [&](size_t bytes) {
    void* p = ws + off;
    off += (bytes + 255) & ~(size_t)255;
    return p;
  };
  const size_t wbytes = (size_t)Cdim * Cdim * 2;
  _Float16* WqT = (_Float16*)carve(wbytes);
  _Float16* WkT = (_Float16*)carve(wbytes);
  _Float16* WvT = (_Float16*)carve(wbytes);
  _Float16* WoT = (_Float16*)carve(wbytes);
  _Float16* qin = (_Float16*)carve((size_t)Bsz * MQ * Cdim * 2);
  _Float16* kin = (_Float16*)carve((size_t)Bsz * MK * Cdim * 2);
  _Float16* vin = (_Float16*)carve((size_t)Bsz * MK * Cdim * 2);
  _Float16* qp  = (_Float16*)carve((size_t)Bsz * MQ * Cdim * 2);
  _Float16* kp  = (_Float16*)carve((size_t)Bsz * MK * Cdim * 2);
  _Float16* vpt = (_Float16*)carve((size_t)Bsz * Cdim * MK * 2);  // [b][c][t]
  _Float16* op  = (_Float16*)carve((size_t)Bsz * MQ * Cdim * 2);

  // stage 0: weight transpose+convert
  int wtN = (Cdim * Cdim + 255) / 256;
  wt_kernel<<<wtN, 256, 0, stream>>>(Wq, WqT);
  wt_kernel<<<wtN, 256, 0, stream>>>(Wk, WkT);
  wt_kernel<<<wtN, 256, 0, stream>>>(Wv, WvT);
  wt_kernel<<<wtN, 256, 0, stream>>>(Wo, WoT);

  // stage 1: depthwise conv + BN -> f16 (zero-padded rows)
  dwbn_q_kernel<<<(Bsz * MQ * Cdim + 255) / 256, 256, 0, stream>>>(
      x, wq_dw, qg, qb, qm, qv, qin);
  dwbn_kv_kernel<<<(Bsz * MK * Cdim + 255) / 256, 256, 0, stream>>>(
      x, wk_dw, kg, kb, km, kv, wv_dw, vg, vb, vm, vv, kin, vin);

  // stage 2: QKV projections (WMMA); V stored transposed for the PV GEMM
  gemm_proj_kernel<<<dim3(MQ / 64, Cdim / 64, Bsz), 128, 0, stream>>>(qin, WqT, qp, MQ, 0);
  gemm_proj_kernel<<<dim3(MK / 64, Cdim / 64, Bsz), 128, 0, stream>>>(kin, WkT, kp, MK, 0);
  gemm_proj_kernel<<<dim3(MK / 64, Cdim / 64, Bsz), 128, 0, stream>>>(vin, WvT, vpt, MK, 1);

  // stage 3: fused softmax attention
  attn_kernel<<<dim3(MQ / 64, NHEAD, Bsz), 128, 0, stream>>>(qp, kp, vpt, op);

  // stage 4: output projection + bias -> f32
  gemm_out_kernel<<<dim3(MQ / 64, Cdim / 64, Bsz), 128, 0, stream>>>(
      op, WoT, bo, (float*)d_out);
}
